// MultiHeadAttentionBlock_72516227826229
// MI455X (gfx1250) — matrix operand order repaired and checked by reading, not compiled, not run on hardware
//
#include <hip/hip_runtime.h>
#include <hip/hip_bf16.h>

// ---------------------------------------------------------------------------
// MHA block for MI455X (gfx1250, wave32, WMMA f32_16x16x32_f16)
//   q/k/v projections: double-buffered LDS GEMM, async global->LDS staging
//   flash attention per (b, h), causal, online softmax, f32 accum
//   out = attn @ Wo^T
// ---------------------------------------------------------------------------

typedef __attribute__((ext_vector_type(16))) _Float16 v16h;
typedef __attribute__((ext_vector_type(8)))  _Float16 h8;
typedef __attribute__((ext_vector_type(8)))  float    v8f;
typedef __attribute__((ext_vector_type(4)))  float    f4;
typedef __attribute__((ext_vector_type(4)))  int      v4i;

#define B_    2
#define S_    2048
#define E_    1024
#define H_    16
#define DH_   64

// ---- CDNA5 async global->LDS copy (ASYNCcnt) ------------------------------
#if defined(__has_builtin)
#if __has_builtin(__builtin_amdgcn_global_load_async_to_lds_b128)
#define HAVE_ASYNC_LDS 1
#endif
#endif
#ifndef HAVE_ASYNC_LDS
#define HAVE_ASYNC_LDS 0
#endif

#if HAVE_ASYNC_LDS
typedef __attribute__((address_space(1))) v4i gv4i;  // global b128 payload
typedef __attribute__((address_space(3))) v4i sv4i;  // LDS b128 payload
__device__ __forceinline__ void async_cp16(void* lds, const void* gmem) {
  __builtin_amdgcn_global_load_async_to_lds_b128(
      (gv4i*)(unsigned long long)gmem, (sv4i*)(unsigned long long)lds, 0, 0);
}
__device__ __forceinline__ void wait_async0() {
#if __has_builtin(__builtin_amdgcn_s_wait_asynccnt)
  __builtin_amdgcn_s_wait_asynccnt(0);
#else
  asm volatile("s_wait_asynccnt 0x0" ::: "memory");
#endif
}
#else
__device__ __forceinline__ void wait_async0() {}
#endif

__device__ __forceinline__ v16h cat16(h8 lo, h8 hi) {
  return __builtin_shufflevector(lo, hi, 0, 1, 2, 3, 4, 5, 6, 7,
                                          8, 9, 10, 11, 12, 13, 14, 15);
}

// 8 contiguous f32 in LDS -> 8 f16
__device__ __forceinline__ h8 cvt8(const float* p) {
  f4 a = *(const f4*)p;
  f4 b = *(const f4*)(p + 4);
  h8 r;
  r[0] = (_Float16)a[0]; r[1] = (_Float16)a[1];
  r[2] = (_Float16)a[2]; r[3] = (_Float16)a[3];
  r[4] = (_Float16)b[0]; r[5] = (_Float16)b[1];
  r[6] = (_Float16)b[2]; r[7] = (_Float16)b[3];
  return r;
}

__device__ __forceinline__ v8f wmma_f16(v16h a, v16h b, v8f c) {
  // (neg_a, A, neg_b, B, c_mod, C, reuse_a, reuse_b)
  return __builtin_amdgcn_wmma_f32_16x16x32_f16(false, a, false, b,
                                                (short)0, c, false, false);
}

// ---------------------------------------------------------------------------
// GEMM: C[M,N] = A[M,K] (f32) @ W[N,K]^T (f32), f16 WMMA, f32 accum.
// Block = 256 threads = 8 waves; block tile 128x64; K tile 32.
// Double-buffered f32 LDS tiles filled by async global->LDS copies; f32->f16
// conversion fused into the fragment build (co-executes with the WMMA pipe).
// ---------------------------------------------------------------------------
template <typename OutT>
__global__ __launch_bounds__(256) void gemm_xwt_kernel(
    const float* __restrict__ A, const float* __restrict__ W,
    OutT* __restrict__ C, int M, int N, int K) {
  __shared__ float As[2][128][36];  // row stride 144B (16B aligned)
  __shared__ float Ws[2][64][36];

  const int tid  = threadIdx.x;
  const int lane = tid & 31;
  const int wave = tid >> 5;
  const int sub  = lane >> 4;   // 16-lane half of the wave
  const int l16  = lane & 15;
  const int rowBase = blockIdx.y * 128;
  const int colBase = blockIdx.x * 64;

  // staging assignment (16B chunks): A 128x32 f32 -> 4 chunks/lane,
  //                                  W  64x32 f32 -> 2 chunks/lane
  const int ar  = tid >> 1;
  const int acb = (tid & 1) * 16;
  const int wr  = tid >> 2;
  const int wcb = (tid & 3) * 8;
  const float* Arow = A + (size_t)(rowBase + ar) * K + acb;
  const float* Wrow = W + (size_t)(colBase + wr) * K + wcb;

  v8f acc[4] = {};
  const int nkt = K / 32;

  // prefetch tile 0 into buffer 0
  {
#if HAVE_ASYNC_LDS
#pragma unroll
    for (int j = 0; j < 16; j += 4) async_cp16(&As[0][ar][acb + j], Arow + j);
#pragma unroll
    for (int j = 0; j < 8; j += 4)  async_cp16(&Ws[0][wr][wcb + j], Wrow + j);
#else
#pragma unroll
    for (int j = 0; j < 16; j += 4) *(f4*)&As[0][ar][acb + j] = *(const f4*)(Arow + j);
#pragma unroll
    for (int j = 0; j < 8; j += 4)  *(f4*)&Ws[0][wr][wcb + j] = *(const f4*)(Wrow + j);
#endif
  }

  for (int kt = 0; kt < nkt; ++kt) {
    wait_async0();
    __syncthreads();  // tile kt resident; all waves done reading buf (kt+1)&1

    if (kt + 1 < nkt) {  // stream next tile while WMMAs run
      const int nb = (kt + 1) & 1;
      const int ko = (kt + 1) * 32;
#if HAVE_ASYNC_LDS
#pragma unroll
      for (int j = 0; j < 16; j += 4) async_cp16(&As[nb][ar][acb + j], Arow + ko + j);
#pragma unroll
      for (int j = 0; j < 8; j += 4)  async_cp16(&Ws[nb][wr][wcb + j], Wrow + ko + j);
#else
#pragma unroll
      for (int j = 0; j < 16; j += 4) *(f4*)&As[nb][ar][acb + j] = *(const f4*)(Arow + ko + j);
#pragma unroll
      for (int j = 0; j < 8; j += 4)  *(f4*)&Ws[nb][wr][wcb + j] = *(const f4*)(Wrow + ko + j);
#endif
    }

    // --- compute from buffer kt&1 -----------------------------------------
    const int cur = kt & 1;
    const int m = wave * 16 + l16;
    // A fragment: K = {8*sub..+7, 16+8*sub..+7}, converted f32->f16
    v16h a = cat16(cvt8(&As[cur][m][8 * sub]), cvt8(&As[cur][m][16 + 8 * sub]));
#pragma unroll
    for (int t = 0; t < 4; ++t) {
      const int n = t * 16 + l16;
      // B fragment: B[k][n] = W[n][k]; K = {8*sub..+7, 16+8*sub..+7}
      v16h bb = cat16(cvt8(&Ws[cur][n][8 * sub]),
                      cvt8(&Ws[cur][n][16 + 8 * sub]));
      acc[t] = wmma_f16(a, bb, acc[t]);
    }
  }

  // --- write C: VGPR r holds M = r + 8*sub, N = l16 (per ISA C layout)
#pragma unroll
  for (int t = 0; t < 4; ++t) {
#pragma unroll
    for (int r = 0; r < 8; ++r) {
      const int row = rowBase + wave * 16 + r + 8 * sub;
      const int col = colBase + t * 16 + l16;
      C[(size_t)row * N + col] = (OutT)acc[t][r];
    }
  }
}

// ---------------------------------------------------------------------------
// Causal flash attention. Grid (S/128, B*H), block 256 = 8 waves.
// Each wave: 16 query rows; loop over 32-key blocks with online softmax.
// Q/K/V are f16 [B*S, E] row-major (head h occupies cols h*64..h*64+63).
// K tile streamed via async global->LDS (1x b128 per lane); V staged
// transposed via scalar scatter (for contiguous P@V B-fragments).
// ---------------------------------------------------------------------------
__global__ __launch_bounds__(256) void flash_attn_kernel(
    const _Float16* __restrict__ Q, const _Float16* __restrict__ Kt,
    const _Float16* __restrict__ Vt, float* __restrict__ Oout) {
  __shared__ _Float16 Ks[32][DH_ + 8];       // [key][dim]   row stride 144B
  __shared__ _Float16 Vts[DH_][32 + 8];      // [dim][key]   row stride 80B
  __shared__ _Float16 Ps[8][16][32 + 8];     // per-wave P tile (C->A transpose)

  const int tid  = threadIdx.x;
  const int lane = tid & 31;
  const int wave = tid >> 5;
  const int sub  = lane >> 4;
  const int l16  = lane & 15;

  const int bh  = blockIdx.y;
  const int bat = bh / H_;
  const int hd  = bh % H_;
  const int wgQ   = blockIdx.x * 128;
  const int qBase = wgQ + wave * 16;
  const size_t baseOff = (size_t)bat * S_ * E_ + (size_t)hd * DH_;

  // --- Q A-fragments: two K-chunks of 32 dims, loaded straight from global
  v16h aq[2];
  {
    const _Float16* qrow = Q + baseOff + (size_t)(qBase + l16) * E_;
#pragma unroll
    for (int c = 0; c < 2; ++c) {
      h8 lo = *(const h8*)(qrow + c * 32 + 8 * sub);
      h8 hi = *(const h8*)(qrow + c * 32 + 16 + 8 * sub);
      aq[c] = cat16(lo, hi);
    }
  }

  v8f o[4] = {};
  float mrun[8], lrun[8];
#pragma unroll
  for (int r = 0; r < 8; ++r) { mrun[r] = -1e30f; lrun[r] = 0.0f; }

  const int kr  = tid >> 3;        // 0..31 (key row for staging)
  const int kcb = (tid & 7) * 8;   // dim chunk (16B)

  const int numKB = wgQ / 32 + 4;  // causal: keys up to wgQ+127
  for (int kb = 0; kb < numKB; ++kb) {
    const int keyBase = kb * 32;
    {
      const _Float16* krow = Kt + baseOff + (size_t)(keyBase + kr) * E_ + kcb;
      const _Float16* vrow = Vt + baseOff + (size_t)(keyBase + kr) * E_ + kcb;
#if HAVE_ASYNC_LDS
      async_cp16(&Ks[kr][kcb], krow);      // K natural layout: direct DMA
#else
      *(h8*)&Ks[kr][kcb] = *(const h8*)krow;
#endif
      h8 vv = *(const h8*)vrow;            // V: transpose scatter into LDS
#pragma unroll
      for (int j = 0; j < 8; ++j) Vts[kcb + j][kr] = vv[j];
    }
    wait_async0();
    __syncthreads();

    const bool active = (keyBase <= qBase + 15);
    if (active) {
      // --- S = Q @ K^T  (two 16-key N-tiles, K-dim 64 = 2 chunks)
      v8f st[2] = {};
#pragma unroll
      for (int c = 0; c < 2; ++c) {
#pragma unroll
        for (int t = 0; t < 2; ++t) {
          h8 lo = *(const h8*)&Ks[t * 16 + l16][c * 32 + 8 * sub];
          h8 hi = *(const h8*)&Ks[t * 16 + l16][c * 32 + 16 + 8 * sub];
          v16h bk = cat16(lo, hi);
          st[t] = wmma_f16(aq[c], bk, st[t]);
        }
      }
      // --- mask + online softmax (row = r + 8*sub spans 16 lanes of `sub`)
      const float scale = 0.125f;  // 1/sqrt(64)
#pragma unroll
      for (int r = 0; r < 8; ++r) {
        const int qi  = qBase + r + 8 * sub;
        const int k0i = keyBase + l16;
        const int k1i = keyBase + 16 + l16;
        float x0 = (k0i > qi) ? -1e30f : st[0][r] * scale;
        float x1 = (k1i > qi) ? -1e30f : st[1][r] * scale;
        float rm = fmaxf(x0, x1);
#pragma unroll
        for (int off = 1; off <= 8; off <<= 1)
          rm = fmaxf(rm, __shfl_xor(rm, off, 32));
        const float nm = fmaxf(mrun[r], rm);
        const float sc = __expf(mrun[r] - nm);
        mrun[r] = nm;
        const float e0 = __expf(x0 - nm);
        const float e1 = __expf(x1 - nm);
        float rs = e0 + e1;
#pragma unroll
        for (int off = 1; off <= 8; off <<= 1)
          rs += __shfl_xor(rs, off, 32);
        lrun[r] = lrun[r] * sc + rs;
#pragma unroll
        for (int t = 0; t < 4; ++t) o[t][r] *= sc;
        // stage P (C layout -> row-major LDS) for A-fragment rebuild
        Ps[wave][r + 8 * sub][l16] = (_Float16)e0;
        Ps[wave][r + 8 * sub][16 + l16] = (_Float16)e1;
      }
    }
    __syncthreads();  // P visible; fences Vts consumption ordering

    if (active) {
      // --- P A-fragment (16x32 over keys)
      h8 plo = *(const h8*)&Ps[wave][l16][8 * sub];
      h8 phi = *(const h8*)&Ps[wave][l16][16 + 8 * sub];
      v16h ap = cat16(plo, phi);
      // --- O += P @ V : 4 dim-tiles of 16; B[k][n] = Vts[n][k], K = {8*sub..+7, 16+8*sub..+7}
#pragma unroll
      for (int t = 0; t < 4; ++t) {
        h8 blo = *(const h8*)&Vts[t * 16 + l16][8 * sub];
        h8 bhi = *(const h8*)&Vts[t * 16 + l16][16 + 8 * sub];
        v16h bv = cat16(blo, bhi);
        o[t] = wmma_f16(ap, bv, o[t]);
      }
    }
    __syncthreads();  // before next block overwrites Ks/Vts
  }

  // --- normalize and write attention output (f32, [B*S, E] row-major)
#pragma unroll
  for (int r = 0; r < 8; ++r) {
    const float inv = 1.0f / lrun[r];
    const int row = qBase + r + 8 * sub;
    const size_t outBase = ((size_t)(bat * S_ + row)) * E_ + hd * DH_;
#pragma unroll
    for (int t = 0; t < 4; ++t)
      Oout[outBase + t * 16 + l16] = o[t][r] * inv;
  }
}

// ---------------------------------------------------------------------------
// Launch: 3 projection GEMMs (f16 out) -> flash attention (f32 out) -> O GEMM
// Workspace: Qb(8MB f16) | Kb(8MB) | Vb(8MB) | attnOut(16MB f32) = 40MB
// ---------------------------------------------------------------------------
extern "C" void kernel_launch(void* const* d_in, const int* in_sizes, int n_in,
                              void* d_out, int out_size, void* d_ws,
                              size_t ws_size, hipStream_t stream) {
  (void)in_sizes; (void)n_in; (void)out_size; (void)ws_size;
  const float* key   = (const float*)d_in[0];
  const float* query = (const float*)d_in[1];
  const float* value = (const float*)d_in[2];
  // d_in[3] = causal mask (tril) -- applied analytically in-kernel
  const float* Wq = (const float*)d_in[4];
  const float* Wk = (const float*)d_in[5];
  const float* Wv = (const float*)d_in[6];
  const float* Wo = (const float*)d_in[7];

  const int M = B_ * S_;  // 4096
  char* ws = (char*)d_ws;
  _Float16* Qb = (_Float16*)(ws);
  _Float16* Kb = (_Float16*)(ws + ((size_t)8 << 20));
  _Float16* Vb = (_Float16*)(ws + ((size_t)16 << 20));
  float*    AO = (float*)(ws + ((size_t)24 << 20));

  dim3 blk(256);
  dim3 ggrid(E_ / 64, M / 128);  // (16, 32)
  gemm_xwt_kernel<_Float16><<<ggrid, blk, 0, stream>>>(query, Wq, Qb, M, E_, E_);
  gemm_xwt_kernel<_Float16><<<ggrid, blk, 0, stream>>>(key,   Wk, Kb, M, E_, E_);
  gemm_xwt_kernel<_Float16><<<ggrid, blk, 0, stream>>>(value, Wv, Vb, M, E_, E_);

  dim3 agrid(S_ / 128, B_ * H_);  // (16, 32)
  flash_attn_kernel<<<agrid, blk, 0, stream>>>(Qb, Kb, Vb, AO);

  gemm_xwt_kernel<float><<<ggrid, blk, 0, stream>>>(AO, Wo, (float*)d_out, M, E_, E_);
}
